// SelfModifyingBlock_53927609368784
// MI455X (gfx1250) — matrix operand order repaired and checked by reading, not compiled, not run on hardware
//
#include <hip/hip_runtime.h>

// ---------------------------------------------------------------------------
// Problem constants (from the reference)
// ---------------------------------------------------------------------------
#define B_      16
#define C_      256
#define HID_    512
#define HW_     3136
#define M_      50176          // B_*HW_
#define CG_     64             // HID_/GROUPS
#define CB_     64             // C_/4 bottleneck
#define LR_     0.01f
#define GN_EPS_ 1e-5f
#define DENOM_  3211264.0f     // inv.sum()*C = 16*784*256  (+1e-8 negligible)
#define NM1_    50175.0f       // n-1 for unbiased channel variance

typedef __bf16 bf16;
typedef __attribute__((ext_vector_type(8)))  __bf16 v8bf;
typedef __attribute__((ext_vector_type(16))) __bf16 v16bf;
typedef __attribute__((ext_vector_type(8)))  float  v8f;

// ---------------------------------------------------------------------------
// Math helpers
// ---------------------------------------------------------------------------
__device__ __forceinline__ float gelu_f(float z) {
    return 0.5f * z * (1.0f + erff(z * 0.70710678118654752f));
}
__device__ __forceinline__ float gelu_grad_f(float z) {
    float cdf = 0.5f * (1.0f + erff(z * 0.70710678118654752f));
    float pdf = 0.39894228040143267f * __expf(-0.5f * z * z);
    return cdf + z * pdf;
}

// ---------------------------------------------------------------------------
// WMMA micro-kernel pieces (CDNA5 wave32, 16x16x32 bf16 -> f32)
// Block tile 128x64, TK=64; 8 waves (4x2), each wave owns 32x32 output:
// 8 WMMAs per barrier round with a0/a1 x b0/b1 fragment reuse.
// ---------------------------------------------------------------------------
#define TM  128
#define TN  64
#define TK  64
#define LDK 72   // LDS pitch in bf16 elems (144 B rows: 16B aligned, conflict-breaking)

__device__ __forceinline__ v8f wmma_bf16(v16bf a, v16bf b, v8f c) {
    return __builtin_amdgcn_wmma_f32_16x16x32_bf16(false, a, false, b,
                                                   (short)0, c, false, false);
}
// A fragment (16x32): rows m = lane&15; kh=8*(lane>>4): K {kh..kh+7, 16+kh..}
__device__ __forceinline__ v16bf load_a_frag(const bf16* lds, int lane) {
    int m  = lane & 15;
    int kh = (lane >> 4) * 8;
    v8bf lo = *(const v8bf*)&lds[m * LDK + kh];
    v8bf hi = *(const v8bf*)&lds[m * LDK + 16 + kh];
    return __builtin_shufflevector(lo, hi, 0,1,2,3,4,5,6,7,8,9,10,11,12,13,14,15);
}
// B fragment (32x16) from BT ([N,K] in LDS): kh=8*(lane>>4): K {kh..kh+7, 16+kh..}
__device__ __forceinline__ v16bf load_b_frag(const bf16* lds, int lane) {
    int n  = lane & 15;
    int kh = (lane >> 4) * 8;
    v8bf lo = *(const v8bf*)&lds[n * LDK + kh];
    v8bf hi = *(const v8bf*)&lds[n * LDK + 16 + kh];
    return __builtin_shufflevector(lo, hi, 0,1,2,3,4,5,6,7,8,9,10,11,12,13,14,15);
}

// C[M,N] = A[M,K] * BT[N,K]^T   (A,BT bf16 row-major, C f32 row-major)
// Requires M%128==0, N%64==0, K%64==0.
__global__ __launch_bounds__(256) void wmma_gemm_nn(
        const bf16* __restrict__ A, const bf16* __restrict__ BT,
        float* __restrict__ Cm, int M, int N, int K) {
    __shared__ __align__(16) bf16 sA[TM * LDK];   // 18432 B
    __shared__ __align__(16) bf16 sB[TN * LDK];   //  9216 B
    const int tid  = threadIdx.x;
    const int lane = tid & 31;
    const int wv   = tid >> 5;
    const int wr   = wv & 3;        // M subtile: 4 x 32 rows
    const int wc   = wv >> 2;       // N subtile: 2 x 32 cols
    const int m0   = blockIdx.y * TM;
    const int n0   = blockIdx.x * TN;
    const int lrA  = tid >> 1;            // 0..127
    const int lcA  = (tid & 1) * 32;      // 0 / 32
    const int lrB  = tid >> 2;            // 0..63
    const int lcB  = (tid & 3) * 16;      // 0,16,32,48

    v8f acc00 = {}, acc01 = {}, acc10 = {}, acc11 = {};
    for (int k0 = 0; k0 < K; k0 += TK) {
        const bf16* gA = &A[(long)(m0 + lrA) * K + k0 + lcA];
#pragma unroll
        for (int j = 0; j < 4; ++j)
            *(v8bf*)&sA[lrA * LDK + lcA + j * 8] = *(const v8bf*)&gA[j * 8];
        const bf16* gB = &BT[(long)(n0 + lrB) * K + k0 + lcB];
#pragma unroll
        for (int j = 0; j < 2; ++j)
            *(v8bf*)&sB[lrB * LDK + lcB + j * 8] = *(const v8bf*)&gB[j * 8];
        if (k0 + TK < K) {   // global_prefetch_b8 of next A tile
            __builtin_prefetch(&gA[TK], 0, 1);
        }
        __syncthreads();
#pragma unroll
        for (int kk = 0; kk < TK; kk += 32) {
            v16bf a0 = load_a_frag(&sA[(wr * 32) * LDK + kk], lane);
            v16bf a1 = load_a_frag(&sA[(wr * 32 + 16) * LDK + kk], lane);
            v16bf b0 = load_b_frag(&sB[(wc * 32) * LDK + kk], lane);
            v16bf b1 = load_b_frag(&sB[(wc * 32 + 16) * LDK + kk], lane);
            acc00 = wmma_bf16(a0, b0, acc00);
            acc01 = wmma_bf16(a0, b1, acc01);
            acc10 = wmma_bf16(a1, b0, acc10);
            acc11 = wmma_bf16(a1, b1, acc11);
        }
        __syncthreads();
    }
    int row = m0 + wr * 32 + ((lane >> 4) * 8);
    int col = n0 + wc * 32 + (lane & 15);
#pragma unroll
    for (int r = 0; r < 8; ++r) {
        Cm[(long)(row + r) * N + col]           = acc00[r];
        Cm[(long)(row + r) * N + col + 16]      = acc01[r];
        Cm[(long)(row + 16 + r) * N + col]      = acc10[r];
        Cm[(long)(row + 16 + r) * N + col + 16] = acc11[r];
    }
}

// C[N1,N2] += A[Mrows,N1]^T * B[Mrows,N2]  (split-K over Mrows, atomic f32 accum)
// Requires N1%128==0, N2%64==0, chunk%64==0.
__global__ __launch_bounds__(256) void wmma_gemm_tn(
        const bf16* __restrict__ A, const bf16* __restrict__ Bm,
        float* __restrict__ Cm, int Mrows, int N1, int N2, int chunk) {
    __shared__ __align__(16) bf16 sAT[TM * LDK];   // [n1][m]
    __shared__ __align__(16) bf16 sBT[TN * LDK];   // [n2][m]
    const int tid  = threadIdx.x;
    const int lane = tid & 31;
    const int wv   = tid >> 5;
    const int wr   = wv & 3;
    const int wc   = wv >> 2;
    const int n1_0 = blockIdx.y * TM;
    const int n2_0 = blockIdx.x * TN;
    const int mstart = blockIdx.z * chunk;
    int mend = mstart + chunk; if (mend > Mrows) mend = Mrows;
    const int sm  = tid >> 2;           // 0..63 (m within K-slab)
    const int snA = (tid & 3) * 32;     // 0..96 by 32
    const int snB = (tid & 3) * 16;     // 0..48 by 16

    v8f acc00 = {}, acc01 = {}, acc10 = {}, acc11 = {};
    for (int m0 = mstart; m0 < mend; m0 += TK) {
#pragma unroll
        for (int j = 0; j < 4; ++j) {   // transpose A slab while staging
            v8bf av = *(const v8bf*)&A[(long)(m0 + sm) * N1 + n1_0 + snA + j * 8];
#pragma unroll
            for (int i = 0; i < 8; ++i) sAT[(snA + j * 8 + i) * LDK + sm] = av[i];
        }
#pragma unroll
        for (int j = 0; j < 2; ++j) {   // transpose B slab while staging
            v8bf bv = *(const v8bf*)&Bm[(long)(m0 + sm) * N2 + n2_0 + snB + j * 8];
#pragma unroll
            for (int i = 0; i < 8; ++i) sBT[(snB + j * 8 + i) * LDK + sm] = bv[i];
        }
        __syncthreads();
#pragma unroll
        for (int kk = 0; kk < TK; kk += 32) {
            v16bf a0 = load_a_frag(&sAT[(wr * 32) * LDK + kk], lane);
            v16bf a1 = load_a_frag(&sAT[(wr * 32 + 16) * LDK + kk], lane);
            v16bf b0 = load_b_frag(&sBT[(wc * 32) * LDK + kk], lane);
            v16bf b1 = load_b_frag(&sBT[(wc * 32 + 16) * LDK + kk], lane);
            acc00 = wmma_bf16(a0, b0, acc00);
            acc01 = wmma_bf16(a0, b1, acc01);
            acc10 = wmma_bf16(a1, b0, acc10);
            acc11 = wmma_bf16(a1, b1, acc11);
        }
        __syncthreads();
    }
    int row = n1_0 + wr * 32 + ((lane >> 4) * 8);
    int col = n2_0 + wc * 32 + (lane & 15);
#pragma unroll
    for (int r = 0; r < 8; ++r) {
        atomicAdd(&Cm[(row + r) * N2 + col],           acc00[r]);
        atomicAdd(&Cm[(row + r) * N2 + col + 16],      acc01[r]);
        atomicAdd(&Cm[(row + 16 + r) * N2 + col],      acc10[r]);
        atomicAdd(&Cm[(row + 16 + r) * N2 + col + 16], acc11[r]);
    }
}

// ---------------------------------------------------------------------------
// Data movement / pointwise / reduction kernels
// ---------------------------------------------------------------------------
__global__ void zero_kernel(float* p, long n) {
    for (long i = (long)blockIdx.x * blockDim.x + threadIdx.x; i < n;
         i += (long)gridDim.x * blockDim.x) p[i] = 0.0f;
}

// x NCHW f32 -> xT [M,C] f32 + bf16   (grid: HW/32, C/32, B)
__global__ __launch_bounds__(256) void transpose_x_kernel(
        const float* __restrict__ x, float* __restrict__ xT, bf16* __restrict__ xTbf) {
    __shared__ float tile[32][33];
    int hw0 = blockIdx.x * 32, c0 = blockIdx.y * 32, b = blockIdx.z;
    int tx = threadIdx.x & 31, ty = threadIdx.x >> 5;
    for (int r = ty; r < 32; r += 8)
        tile[r][tx] = x[((long)(b * C_ + c0 + r)) * HW_ + hw0 + tx];
    __syncthreads();
    for (int r = ty; r < 32; r += 8) {
        long m = (long)b * HW_ + hw0 + r;
        float v = tile[tx][r];
        xT[m * C_ + c0 + tx]   = v;
        xTbf[m * C_ + c0 + tx] = (bf16)v;
    }
}

// One-time weight prep: f32 masters + bf16 (and transposed bf16) copies
__global__ void weights_init_kernel(
        const float* __restrict__ c1, const float* __restrict__ g_in,
        const float* __restrict__ b_in, const float* __restrict__ c2,
        const float* __restrict__ rw1, const float* __restrict__ rw2,
        float* W1, bf16* W1bf, float* gamma, float* beta,
        float* W2, bf16* W2bf, bf16* W2Tbf,
        bf16* Rw1bf, bf16* Rw1Tbf, bf16* Rw2bf, bf16* Rw2Tbf) {
    int i = blockIdx.x * blockDim.x + threadIdx.x;
    if (i < HID_ * C_) {
        float v = c1[i]; W1[i] = v; W1bf[i] = (bf16)v;        // [HID,C] = BT for G1
        float w = c2[i]; W2[i] = w; W2bf[i] = (bf16)w;        // [C,HID] = BT for G2
        int c = i >> 9, o = i & 511;                          // W2 index (c,o)
        W2Tbf[o * C_ + c] = (bf16)w;                          // [HID,C] BT for dL/da
    }
    if (i < CB_ * C_) {
        float v = rw1[i]; Rw1bf[i] = (bf16)v;                 // [CB,C]
        int o = i >> 8, c = i & 255;
        Rw1Tbf[c * CB_ + o] = (bf16)v;                        // [C,CB]
        float w = rw2[i]; Rw2bf[i] = (bf16)w;                 // [C,CB] (row-major)
        int cc = i >> 6, oo = i & 63;                         // rw2 index (c,o)
        Rw2Tbf[oo * C_ + cc] = (bf16)w;                       // [CB,C]
    }
    if (i < HID_) { gamma[i] = g_in[i]; beta[i] = b_in[i]; }
}

// GroupNorm stats per (b,g)  (128 blocks)
__global__ __launch_bounds__(256) void gn_stats_kernel(
        const float* __restrict__ h1, float* __restrict__ mu, float* __restrict__ rstd) {
    int bg = blockIdx.x, b = bg >> 3, g = bg & 7;
    long base = (long)b * HW_ * HID_ + g * CG_;
    float s = 0.f, s2 = 0.f;
    for (int i = threadIdx.x; i < HW_ * CG_; i += 256) {
        int hw = i >> 6, ol = i & 63;
        float v = h1[base + (long)hw * HID_ + ol];
        s += v; s2 += v * v;
    }
    __shared__ float r1[256], r2[256];
    r1[threadIdx.x] = s; r2[threadIdx.x] = s2; __syncthreads();
    for (int st = 128; st > 0; st >>= 1) {
        if (threadIdx.x < st) { r1[threadIdx.x] += r1[threadIdx.x + st];
                                r2[threadIdx.x] += r2[threadIdx.x + st]; }
        __syncthreads();
    }
    if (threadIdx.x == 0) {
        float inv = 1.0f / (float)(HW_ * CG_);
        float m = r1[0] * inv;
        float v = r2[0] * inv - m * m;
        mu[bg] = m; rstd[bg] = rsqrtf(v + GN_EPS_);
    }
}

// a = gelu(GN(h1)*gamma+beta) -> bf16
__global__ void act_kernel(const float* __restrict__ h1, const float* __restrict__ mu,
                           const float* __restrict__ rstd, const float* __restrict__ gamma,
                           const float* __restrict__ beta, bf16* __restrict__ a_bf) {
    long n = (long)M_ * HID_;
    for (long i = (long)blockIdx.x * blockDim.x + threadIdx.x; i < n;
         i += (long)gridDim.x * blockDim.x) {
        int o = (int)(i & (HID_ - 1)); long m = i >> 9;
        int bg = (int)(m / HW_) * 8 + (o >> 6);
        float hn = (h1[i] - mu[bg]) * rstd[bg];
        a_bf[i] = (bf16)gelu_f(hn * gamma[o] + beta[o]);
    }
}

// y = m + feat (in place), my_bf = y*mask
__global__ void y_my_kernel(float* __restrict__ ybuf, const float* __restrict__ xT,
                            const unsigned char* __restrict__ mask, bf16* __restrict__ my_bf) {
    long n = (long)M_ * C_;
    for (long i = (long)blockIdx.x * blockDim.x + threadIdx.x; i < n;
         i += (long)gridDim.x * blockDim.x) {
        long m = i >> 8;
        float y = ybuf[i] + xT[i];
        ybuf[i] = y;
        my_bf[i] = (bf16)(mask[m] ? y : 0.0f);
    }
}

// per-channel sum / sum-sq of y (256 blocks x 196 rows, thread = channel)
__global__ __launch_bounds__(256) void colsum_kernel(const float* __restrict__ y,
                                                     float* __restrict__ cs, float* __restrict__ cs2) {
    int c = threadIdx.x;
    long r0 = (long)blockIdx.x * 196;
    float s = 0.f, s2 = 0.f;
    for (int r = 0; r < 196; ++r) {
        float v = y[(r0 + r) * C_ + c]; s += v; s2 += v * v;
    }
    atomicAdd(&cs[c], s); atomicAdd(&cs2[c], s2);
}

// consistency-loss coefficients for dL/dy
__global__ void consist_kernel(const float* __restrict__ cs, const float* __restrict__ cs2,
                               const float* __restrict__ rm, const float* __restrict__ rv,
                               float* __restrict__ cm, float* __restrict__ alpha,
                               float* __restrict__ betac) {
    int c = threadIdx.x;
    float n = (float)M_;
    float mean = cs[c] / n;
    float cv = (cs2[c] - n * mean * mean) / NM1_;
    cm[c] = mean;
    alpha[c] = 0.1f * 2.0f * (mean - rm[c]) / ((float)C_ * n);
    float rvp = rv[c] + 1e-8f;
    betac[c] = 0.1f * (2.0f * (cv / rvp - 1.0f) / (rvp * (float)C_)) * (2.0f / NM1_);
}

__global__ void gelu_t_kernel(const float* __restrict__ t, bf16* __restrict__ g1) {
    long n = (long)M_ * CB_;
    for (long i = (long)blockIdx.x * blockDim.x + threadIdx.x; i < n;
         i += (long)gridDim.x * blockDim.x) g1[i] = (bf16)gelu_f(t[i]);
}

// drec = (2/D)*e*inv ; dy = -drec + alpha + betac*(y-cm)
__global__ void dy_drec_kernel(const float* __restrict__ rec, const float* __restrict__ y,
                               const unsigned char* __restrict__ mask,
                               const float* __restrict__ cm, const float* __restrict__ alpha,
                               const float* __restrict__ betac,
                               bf16* __restrict__ drec_bf, float* __restrict__ dy) {
    long n = (long)M_ * C_;
    for (long i = (long)blockIdx.x * blockDim.x + threadIdx.x; i < n;
         i += (long)gridDim.x * blockDim.x) {
        int c = (int)(i & (C_ - 1)); long m = i >> 8;
        float inv = mask[m] ? 0.0f : 1.0f;
        float dr = (2.0f / DENOM_) * (rec[i] - y[i]) * inv;
        drec_bf[i] = (bf16)dr;
        dy[i] = -dr + alpha[c] + betac[c] * (y[i] - cm[c]);
    }
}

__global__ void dt_kernel(const float* __restrict__ dg1, const float* __restrict__ t,
                          bf16* __restrict__ dt_bf) {
    long n = (long)M_ * CB_;
    for (long i = (long)blockIdx.x * blockDim.x + threadIdx.x; i < n;
         i += (long)gridDim.x * blockDim.x)
        dt_bf[i] = (bf16)(dg1[i] * gelu_grad_f(t[i]));
}

// dm = dy + dmy*mask  -> bf16
__global__ void dm_kernel(const float* __restrict__ dy, const float* __restrict__ dmy,
                          const unsigned char* __restrict__ mask, bf16* __restrict__ dm_bf) {
    long n = (long)M_ * C_;
    for (long i = (long)blockIdx.x * blockDim.x + threadIdx.x; i < n;
         i += (long)gridDim.x * blockDim.x) {
        long m = i >> 8;
        dm_bf[i] = (bf16)(dy[i] + (mask[m] ? dmy[i] : 0.0f));
    }
}

// GN backward reductions per (b,g): dgamma/dbeta (atomic) + Sg1/Sg2
__global__ __launch_bounds__(256) void gn_bwd_reduce(
        const float* __restrict__ h1, const float* __restrict__ da,
        const float* __restrict__ mu, const float* __restrict__ rstd,
        const float* __restrict__ gamma, const float* __restrict__ beta,
        float* __restrict__ dgamma, float* __restrict__ dbeta,
        float* __restrict__ Sg1, float* __restrict__ Sg2) {
    int bg = blockIdx.x, b = bg >> 3, g = bg & 7;
    long base = (long)b * HW_ * HID_ + g * CG_;
    int ol = threadIdx.x & 63, sub = threadIdx.x >> 6;
    int oc = g * CG_ + ol;
    float mu_ = mu[bg], rs_ = rstd[bg];
    float gam = gamma[oc], bet = beta[oc];
    float dg = 0.f, db = 0.f, s1 = 0.f, s2 = 0.f;
    for (int hw = sub; hw < HW_; hw += 4) {
        long idx = base + (long)hw * HID_ + ol;
        float hn = (h1[idx] - mu_) * rs_;
        float dhgn = da[idx] * gelu_grad_f(hn * gam + bet);
        dg += dhgn * hn; db += dhgn;
        float dhn = dhgn * gam;
        s1 += dhn; s2 += dhn * hn;
    }
    __shared__ float r1[256], r2[256];
    r1[threadIdx.x] = dg; r2[threadIdx.x] = db; __syncthreads();
    if (sub == 0) {
        atomicAdd(&dgamma[oc], r1[ol] + r1[64 + ol] + r1[128 + ol] + r1[192 + ol]);
        atomicAdd(&dbeta[oc],  r2[ol] + r2[64 + ol] + r2[128 + ol] + r2[192 + ol]);
    }
    __syncthreads();
    r1[threadIdx.x] = s1; r2[threadIdx.x] = s2; __syncthreads();
    for (int st = 128; st > 0; st >>= 1) {
        if (threadIdx.x < st) { r1[threadIdx.x] += r1[threadIdx.x + st];
                                r2[threadIdx.x] += r2[threadIdx.x + st]; }
        __syncthreads();
    }
    if (threadIdx.x == 0) { Sg1[bg] = r1[0]; Sg2[bg] = r2[0]; }
}

// dh1 = rstd*(dhn - (Sg1 + hn*Sg2)/Ng) -> bf16
__global__ void gn_dh1_kernel(const float* __restrict__ h1, const float* __restrict__ da,
                              const float* __restrict__ mu, const float* __restrict__ rstd,
                              const float* __restrict__ gamma, const float* __restrict__ beta,
                              const float* __restrict__ Sg1, const float* __restrict__ Sg2,
                              bf16* __restrict__ dh1_bf) {
    const float invN = 1.0f / (float)(CG_ * HW_);
    long n = (long)M_ * HID_;
    for (long i = (long)blockIdx.x * blockDim.x + threadIdx.x; i < n;
         i += (long)gridDim.x * blockDim.x) {
        int o = (int)(i & (HID_ - 1)); long m = i >> 9;
        int bg = (int)(m / HW_) * 8 + (o >> 6);
        float rs_ = rstd[bg];
        float hn = (h1[i] - mu[bg]) * rs_;
        float dhn = da[i] * gelu_grad_f(hn * gamma[o] + beta[o]) * gamma[o];
        dh1_bf[i] = (bf16)(rs_ * (dhn - (Sg1[bg] + hn * Sg2[bg]) * invN));
    }
}

// SGD update + refresh bf16 copies (131072 threads)
__global__ void sgd_update_kernel(float* W1, bf16* W1bf, float* W2, bf16* W2bf, bf16* W2Tbf,
                                  float* gamma, float* beta,
                                  const float* dW1, const float* dW2,
                                  const float* dgam, const float* dbet) {
    int i = blockIdx.x * blockDim.x + threadIdx.x;
    if (i < HID_ * C_) {
        float w1 = W1[i] - LR_ * dW1[i]; W1[i] = w1; W1bf[i] = (bf16)w1;
        float w2 = W2[i] - LR_ * dW2[i]; W2[i] = w2; W2bf[i] = (bf16)w2;
        int c = i >> 9, o = i & 511;
        W2Tbf[o * C_ + c] = (bf16)w2;
    }
    if (i < HID_) { gamma[i] -= LR_ * dgam[i]; beta[i] -= LR_ * dbet[i]; }
}

// pooled[b,c] = mean_hw x  (4096 blocks)
__global__ __launch_bounds__(256) void pooled_kernel(const float* __restrict__ x,
                                                     float* __restrict__ pooled) {
    int bc = blockIdx.x;
    const float* p = x + (long)bc * HW_;
    float s = 0.f;
    for (int i = threadIdx.x; i < HW_; i += 256) s += p[i];
    __shared__ float sh[256];
    sh[threadIdx.x] = s; __syncthreads();
    for (int st = 128; st > 0; st >>= 1) {
        if (threadIdx.x < st) sh[threadIdx.x] += sh[threadIdx.x + st];
        __syncthreads();
    }
    if (threadIdx.x == 0) pooled[bc] = sh[0] * (1.0f / (float)HW_);
}

// gate MLP (16 blocks x 64 threads)
__global__ __launch_bounds__(64) void gate_kernel(const float* __restrict__ pooled,
        const float* __restrict__ gw1, const float* __restrict__ gb1,
        const float* __restrict__ gw2, const float* __restrict__ gb2,
        float* __restrict__ gate) {
    int b = blockIdx.x, o = threadIdx.x;
    float s = gb1[o];
    for (int c = 0; c < C_; ++c) s += pooled[b * C_ + c] * gw1[o * C_ + c];
    __shared__ float sh[64];
    sh[o] = gelu_f(s) * gw2[o]; __syncthreads();
    for (int st = 32; st > 0; st >>= 1) {
        if (o < st) sh[o] += sh[o + st];
        __syncthreads();
    }
    if (o == 0) gate[b] = 1.0f / (1.0f + __expf(-(sh[0] + gb2[0])));
}

// out[b,c,hw] = x + rs*gate[b]*mod  (tiled [M,C] -> NCHW transpose)
__global__ __launch_bounds__(256) void final_out_kernel(
        const float* __restrict__ x, const float* __restrict__ mod,
        const float* __restrict__ gate, const float* __restrict__ rs,
        float* __restrict__ out) {
    __shared__ float tile[32][33];
    int hw0 = blockIdx.x * 32, c0 = blockIdx.y * 32, b = blockIdx.z;
    int tx = threadIdx.x & 31, ty = threadIdx.x >> 5;
    for (int r = ty; r < 32; r += 8)
        tile[r][tx] = mod[((long)b * HW_ + hw0 + r) * C_ + c0 + tx];
    __syncthreads();
    float sc = rs[0] * gate[b];
    for (int r = ty; r < 32; r += 8) {
        long oi = ((long)(b * C_ + c0 + r)) * HW_ + hw0 + tx;
        out[oi] = x[oi] + sc * tile[tx][r];
    }
}

// ---------------------------------------------------------------------------
// Host orchestration
// ---------------------------------------------------------------------------
extern "C" void kernel_launch(void* const* d_in, const int* in_sizes, int n_in,
                              void* d_out, int out_size, void* d_ws, size_t ws_size,
                              hipStream_t stream) {
    const float* x        = (const float*)d_in[0];
    const float* conv1_w  = (const float*)d_in[1];
    const float* gn_gamma = (const float*)d_in[2];
    const float* gn_beta  = (const float*)d_in[3];
    const float* conv2_w  = (const float*)d_in[4];
    const float* recon_w1 = (const float*)d_in[5];
    const float* recon_w2 = (const float*)d_in[6];
    const float* gate_w1  = (const float*)d_in[7];
    const float* gate_b1  = (const float*)d_in[8];
    const float* gate_w2  = (const float*)d_in[9];
    const float* gate_b2  = (const float*)d_in[10];
    const float* res_scale= (const float*)d_in[11];
    const float* run_mean = (const float*)d_in[12];
    const float* run_var  = (const float*)d_in[13];
    const unsigned char* masks = (const unsigned char*)d_in[14];

    char* ws = (char*)d_ws;
    size_t off = 0;
    auto alloc = [&](size_t bytes) -> void* {
        void* p = ws + off;
        off = (off + bytes + 255) & ~(size_t)255;
        return p;
    };
    float* xT     = (float*)alloc((size_t)M_ * C_ * 4);
    bf16*  xTbf   = (bf16*) alloc((size_t)M_ * C_ * 2);
    float* h1     = (float*)alloc((size_t)M_ * HID_ * 4);
    bf16*  a_bf   = (bf16*) alloc((size_t)M_ * HID_ * 2);
    float* ybuf   = (float*)alloc((size_t)M_ * C_ * 4);   // m, then y (or final mod)
    bf16*  my_bf  = (bf16*) alloc((size_t)M_ * C_ * 2);
    float* tbuf   = (float*)alloc((size_t)M_ * CB_ * 4);
    bf16*  g1_bf  = (bf16*) alloc((size_t)M_ * CB_ * 2);
    float* recbuf = (float*)alloc((size_t)M_ * C_ * 4);   // rec, then dmy
    float* dybuf  = (float*)alloc((size_t)M_ * C_ * 4);
    bf16*  drec_bf= (bf16*) alloc((size_t)M_ * C_ * 2);
    float* dg1    = (float*)alloc((size_t)M_ * CB_ * 4);
    bf16*  dt_bf  = (bf16*) alloc((size_t)M_ * CB_ * 2);
    bf16*  dm_bf  = (bf16*) alloc((size_t)M_ * C_ * 2);
    float* dabuf  = (float*)alloc((size_t)M_ * HID_ * 4);
    bf16*  dh1_bf = (bf16*) alloc((size_t)M_ * HID_ * 2);
    // parameters (f32 masters + bf16 copies)
    float* W1     = (float*)alloc(HID_ * C_ * 4);
    float* W2     = (float*)alloc(HID_ * C_ * 4);
    float* gammaP = (float*)alloc(HID_ * 4);
    float* betaP  = (float*)alloc(HID_ * 4);
    bf16*  W1bf   = (bf16*) alloc(HID_ * C_ * 2);
    bf16*  W2bf   = (bf16*) alloc(HID_ * C_ * 2);
    bf16*  W2Tbf  = (bf16*) alloc(HID_ * C_ * 2);
    bf16*  Rw1bf  = (bf16*) alloc(CB_ * C_ * 2);
    bf16*  Rw1Tbf = (bf16*) alloc(CB_ * C_ * 2);
    bf16*  Rw2bf  = (bf16*) alloc(CB_ * C_ * 2);
    bf16*  Rw2Tbf = (bf16*) alloc(CB_ * C_ * 2);
    // zeroed-per-step accumulator region (contiguous)
    const long ZERO_N = 131072L + 131072L + 512 + 512 + 128 + 128 + 256 + 256;
    float* gradbase = (float*)alloc((size_t)ZERO_N * 4);
    float* dW1    = gradbase;
    float* dW2    = dW1 + 131072;
    float* dgamma = dW2 + 131072;
    float* dbeta  = dgamma + 512;
    float* Sg1    = dbeta + 512;
    float* Sg2    = Sg1 + 128;
    float* colsum = Sg2 + 128;
    float* colsum2= colsum + 256;
    // small stats
    float* mu     = (float*)alloc(128 * 4);
    float* rstd   = (float*)alloc(128 * 4);
    float* cm     = (float*)alloc(256 * 4);
    float* alphaC = (float*)alloc(256 * 4);
    float* betaC  = (float*)alloc(256 * 4);
    float* pooled = (float*)alloc(B_ * C_ * 4);
    float* gate   = (float*)alloc(B_ * 4);

    const dim3 blk256(256);
    const dim3 gT(HW_ / 32, C_ / 32, B_);

    // ---- setup ----
    transpose_x_kernel<<<gT, blk256, 0, stream>>>(x, xT, xTbf);
    weights_init_kernel<<<512, blk256, 0, stream>>>(conv1_w, gn_gamma, gn_beta, conv2_w,
                                                    recon_w1, recon_w2,
                                                    W1, W1bf, gammaP, betaP,
                                                    W2, W2bf, W2Tbf,
                                                    Rw1bf, Rw1Tbf, Rw2bf, Rw2Tbf);
    pooled_kernel<<<B_ * C_, blk256, 0, stream>>>(x, pooled);
    gate_kernel<<<B_, 64, 0, stream>>>(pooled, gate_w1, gate_b1, gate_w2, gate_b2, gate);

    // ---- inner TTT loop: 2 gradient steps ----
    for (int step = 0; step < 2; ++step) {
        const unsigned char* mstep = masks + (size_t)step * B_ * HW_;
        zero_kernel<<<256, blk256, 0, stream>>>(gradbase, ZERO_N);
        // forward modifier
        wmma_gemm_nn<<<dim3(HID_/TN, M_/TM), blk256, 0, stream>>>(xTbf, W1bf, h1, M_, HID_, C_);
        gn_stats_kernel<<<128, blk256, 0, stream>>>(h1, mu, rstd);
        act_kernel<<<2048, blk256, 0, stream>>>(h1, mu, rstd, gammaP, betaP, a_bf);
        wmma_gemm_nn<<<dim3(C_/TN, M_/TM), blk256, 0, stream>>>(a_bf, W2bf, ybuf, M_, C_, HID_);
        // surprise forward
        y_my_kernel<<<2048, blk256, 0, stream>>>(ybuf, xT, mstep, my_bf);
        colsum_kernel<<<256, blk256, 0, stream>>>(ybuf, colsum, colsum2);
        consist_kernel<<<1, blk256, 0, stream>>>(colsum, colsum2, run_mean, run_var,
                                                 cm, alphaC, betaC);
        wmma_gemm_nn<<<dim3(CB_/TN, M_/TM), blk256, 0, stream>>>(my_bf, Rw1bf, tbuf, M_, CB_, C_);
        gelu_t_kernel<<<2048, blk256, 0, stream>>>(tbuf, g1_bf);
        wmma_gemm_nn<<<dim3(C_/TN, M_/TM), blk256, 0, stream>>>(g1_bf, Rw2bf, recbuf, M_, C_, CB_);
        // backward to dm
        dy_drec_kernel<<<2048, blk256, 0, stream>>>(recbuf, ybuf, mstep, cm, alphaC, betaC,
                                                    drec_bf, dybuf);
        wmma_gemm_nn<<<dim3(CB_/TN, M_/TM), blk256, 0, stream>>>(drec_bf, Rw2Tbf, dg1, M_, CB_, C_);
        dt_kernel<<<2048, blk256, 0, stream>>>(dg1, tbuf, dt_bf);
        wmma_gemm_nn<<<dim3(C_/TN, M_/TM), blk256, 0, stream>>>(dt_bf, Rw1Tbf, recbuf, M_, C_, CB_);
        dm_kernel<<<2048, blk256, 0, stream>>>(dybuf, recbuf, mstep, dm_bf);
        // backward through modifier
        wmma_gemm_nn<<<dim3(HID_/TN, M_/TM), blk256, 0, stream>>>(dm_bf, W2Tbf, dabuf, M_, HID_, C_);
        gn_bwd_reduce<<<128, blk256, 0, stream>>>(h1, dabuf, mu, rstd, gammaP, betaP,
                                                  dgamma, dbeta, Sg1, Sg2);
        gn_dh1_kernel<<<2048, blk256, 0, stream>>>(h1, dabuf, mu, rstd, gammaP, betaP,
                                                   Sg1, Sg2, dh1_bf);
        // weight gradients (split-K WMMA TN, atomic f32 accumulate)
        wmma_gemm_tn<<<dim3(HID_/TN, C_/TM, 16), blk256, 0, stream>>>(dm_bf, a_bf, dW2,
                                                                      M_, C_, HID_, 3136);
        wmma_gemm_tn<<<dim3(C_/TN, HID_/TM, 16), blk256, 0, stream>>>(dh1_bf, xTbf, dW1,
                                                                      M_, HID_, C_, 3136);
        sgd_update_kernel<<<512, blk256, 0, stream>>>(W1, W1bf, W2, W2bf, W2Tbf,
                                                      gammaP, betaP, dW1, dW2, dgamma, dbeta);
    }

    // ---- final modifier forward with updated params + gated residual ----
    wmma_gemm_nn<<<dim3(HID_/TN, M_/TM), blk256, 0, stream>>>(xTbf, W1bf, h1, M_, HID_, C_);
    gn_stats_kernel<<<128, blk256, 0, stream>>>(h1, mu, rstd);
    act_kernel<<<2048, blk256, 0, stream>>>(h1, mu, rstd, gammaP, betaP, a_bf);
    wmma_gemm_nn<<<dim3(C_/TN, M_/TM), blk256, 0, stream>>>(a_bf, W2bf, ybuf, M_, C_, HID_);
    final_out_kernel<<<gT, blk256, 0, stream>>>(x, ybuf, gate, res_scale, (float*)d_out);

    (void)in_sizes; (void)n_in; (void)out_size; (void)ws_size;
}
